// cross_perceptron3_54477365183191
// MI455X (gfx1250) — matrix operand order repaired and checked by reading, not compiled, not run on hardware
//
#include <hip/hip_runtime.h>

// ---------------------------------------------------------------------------
// CDNA5 (gfx1250) wave32 WMMA implementation of the cross-attention block.
// All GEMMs run through v_wmma_f32_16x16x32_f16 (f16 operands, f32 accum).
// Fragments load as ds_load_b128 pairs; QKV/FFN GEMMs use 32-row M tiles so
// each staged B tile feeds 2x the WMMAs (B fragment reused across M tiles).
// ---------------------------------------------------------------------------

typedef __attribute__((ext_vector_type(16))) _Float16 v16h;
typedef __attribute__((ext_vector_type(8)))  float    v8f;

static constexpr int NROW = 256 * 49;   // 12544 total tokens
static constexpr int DIMC = 512;

// LDS strides (halves). All keep 16-byte alignment for uint4 access and are
// NOT multiples of 128 halves (64 dwords) so lanes don't alias to one bank.
static constexpr int LDA  = 40;   // A tile row stride  (Mx32 tile)
static constexpr int LDW  = 40;   // W tile row stride  (512 x 32 tile)
static constexpr int LDQK = 264;  // Q/K LDS row stride (64 x 256 tile)
static constexpr int LDVT = 72;   // V^T LDS row stride (256 x 64 tile)
static constexpr int LDP  = 72;   // P LDS row stride   (64 x 64 tile)
static constexpr int LDS_S = 65;  // S (float) row stride

union V16 { uint4 q[2]; v16h v; };
union H8  { uint4 q; _Float16 h[8]; };

__device__ inline v8f v8zero() {
  v8f z = {0.f, 0.f, 0.f, 0.f, 0.f, 0.f, 0.f, 0.f};
  return z;
}

__device__ inline v8f wmma16(v16h a, v16h b, v8f c) {
  return __builtin_amdgcn_wmma_f32_16x16x32_f16(
      /*neg_a=*/false, a, /*neg_b=*/false, b,
      /*c_mod=*/(short)0, c, /*reuse_a=*/false, /*reuse_b=*/false);
}

// A tile: 16(m) x 32(k), row-major, stride ld halves (16B-aligned rows).
// ISA lane layout: lanes 0-15 -> K {0..7,16..23}, lanes 16-31 -> K {8..15,24..31}
__device__ inline v16h load_A16x32_v(const _Float16* A, int lane, int ld) {
  int row = lane & 15;
  int k0  = (lane & 16) ? 8 : 0;
  const _Float16* ar = A + row * ld + k0;
  V16 u;
  u.q[0] = *(const uint4*)(ar);
  u.q[1] = *(const uint4*)(ar + 16);
  return u.v;
}

// B tile from K-contiguous (transposed) storage: memory holds [n][k].
__device__ inline v16h load_Bk16(const _Float16* Bt, int lane, int ld) {
  int n  = lane & 15;
  int k0 = (lane & 16) ? 8 : 0;
  const _Float16* p = Bt + n * ld + k0;
  V16 u;
  u.q[0] = *(const uint4*)(p);
  u.q[1] = *(const uint4*)(p + 16);
  return u.v;
}

// ---------------------------------------------------------------------------
// Weight pre-pass: Wt[n*K + k] = (f16) W[k*N + n]   (one-time per launch)
// ---------------------------------------------------------------------------
__global__ __launch_bounds__(256) void wtrans_kernel(const float* __restrict__ W,
                                                     _Float16* __restrict__ Wt,
                                                     int K, int N) {
  int idx = blockIdx.x * 256 + threadIdx.x;
  if (idx < K * N) {
    int n = idx / K, k = idx % K;          // consecutive threads: contiguous k
    Wt[idx] = (_Float16)W[(size_t)k * N + n];
  }
}

// ---------------------------------------------------------------------------
// Shared GEMM mainloop: block computes MT*16 rows x 512 cols.
// A: f16 [*, KTOT] row-major.  Wt: f16 [512][KTOT] (N-major, K-contiguous).
// acc: flattened v8f[MT*4] (m*4 + nt).
// ---------------------------------------------------------------------------
template <int KTOT, int MT>
__device__ inline void gemm_mainloop(const _Float16* __restrict__ Ag, int rowBase,
                                     const _Float16* __restrict__ Wt,
                                     _Float16* Alds, _Float16* Wlds,
                                     v8f* acc, int tid) {
  int wave = tid >> 5, lane = tid & 31;
  for (int kc = 0; kc < KTOT; kc += 32) {
#pragma unroll
    for (int m = 0; m < MT; ++m) {  // A tiles: MT*16 x 32 halves, dword copies
      int row = (tid >> 4) + m * 16;
      int kk2 = (tid & 15) * 2;
      *(unsigned int*)(Alds + row * LDA + kk2) =
          *(const unsigned int*)(Ag + (size_t)(rowBase + row) * KTOT + kc + kk2);
    }
    for (int i = tid; i < 2048; i += 256) {  // W tile: 512 x 32 halves, b128
      int nrow = i >> 2, q8 = (i & 3) * 8;
      *(uint4*)(Wlds + nrow * LDW + q8) =
          *(const uint4*)(Wt + (size_t)nrow * KTOT + kc + q8);
    }
    __syncthreads();
    v16h a[MT];
#pragma unroll
    for (int m = 0; m < MT; ++m) a[m] = load_A16x32_v(Alds + m * 16 * LDA, lane, LDA);
#pragma unroll
    for (int nt = 0; nt < 4; ++nt) {
      v16h b = load_Bk16(Wlds + (wave * 64 + nt * 16) * LDW, lane, LDW);
#pragma unroll
      for (int m = 0; m < MT; ++m) acc[m * 4 + nt] = wmma16(a[m], b, acc[m * 4 + nt]);
    }
    __syncthreads();
  }
}

// LayerNorm epilogue over 16 rows x 512 cols held in LDS (fp32), writes f16.
__device__ inline void ln_store(const float* Clds, float* red, int tid, int rowBase,
                                const float* __restrict__ g, const float* __restrict__ bb,
                                _Float16* __restrict__ outh) {
  int row = tid >> 4, seg = tid & 15;
  float s = 0.f, s2 = 0.f;
#pragma unroll 4
  for (int j = 0; j < 32; ++j) {
    float v = Clds[row * 512 + seg * 32 + j];
    s += v; s2 += v * v;
  }
  red[row * 16 + seg] = s;
  red[256 + row * 16 + seg] = s2;
  __syncthreads();
  if (seg == 0) {
    float a = 0.f, c = 0.f;
    for (int j = 0; j < 16; ++j) { a += red[row * 16 + j]; c += red[256 + row * 16 + j]; }
    red[row * 16] = a;
    red[256 + row * 16] = c;
  }
  __syncthreads();
  for (int i = tid; i < 16 * 512; i += 256) {
    int rr = i >> 9, cc = i & 511;
    float m   = red[rr * 16] * (1.f / 512.f);
    float var = red[256 + rr * 16] * (1.f / 512.f) - m * m;
    float inv = rsqrtf(var + 1e-5f);
    outh[(size_t)(rowBase + rr) * 512 + cc] = (_Float16)((Clds[i] - m) * inv * g[cc] + bb[cc]);
  }
}

// ---------------------------------------------------------------------------
// Kernel 1: patchify + linear embed + bias + LayerNorm (16-row blocks).
// ---------------------------------------------------------------------------
template <int CIN, int P, bool WRITE_RAW>
__global__ __launch_bounds__(256) void embed_ln_kernel(
    const float* __restrict__ src, const _Float16* __restrict__ Wt,
    const float* __restrict__ bias, const float* __restrict__ g,
    const float* __restrict__ bb, _Float16* __restrict__ outh,
    float* __restrict__ rawOut) {
  constexpr int KTOT = P * P * CIN;
  constexpr int H = 7 * P;
  __shared__ __align__(16) _Float16 Alds[16 * LDA];
  __shared__ __align__(16) _Float16 Wlds[512 * LDW];
  __shared__ float Clds[16 * 512];
  __shared__ float red[512];
  int tid = threadIdx.x;
  int rowBase = blockIdx.x * 16;
  int wave = tid >> 5, lane = tid & 31;
  v8f acc[4];
#pragma unroll
  for (int i = 0; i < 4; ++i) acc[i] = v8zero();

  for (int kc = 0; kc < KTOT; kc += 32) {
    for (int i = tid; i < 512; i += 256) {  // patchify gather into A tile
      int row = i >> 5, kk = i & 31;
      int r = rowBase + row;
      int bb2 = r / 49, t = r % 49;
      int hh = t / 7, wv = t % 7;
      int f = kc + kk;
      int p1  = f / (P * CIN);
      int rem = f % (P * CIN);
      int p2  = rem / CIN;
      int c   = rem % CIN;
      size_t addr = (((size_t)bb2 * CIN + c) * H + (hh * P + p1)) * H + (wv * P + p2);
      Alds[row * LDA + kk] = (_Float16)src[addr];
    }
    for (int i = tid; i < 2048; i += 256) {  // W tile (pre-transposed f16)
      int nrow = i >> 2, q8 = (i & 3) * 8;
      *(uint4*)(Wlds + nrow * LDW + q8) =
          *(const uint4*)(Wt + (size_t)nrow * KTOT + kc + q8);
    }
    __syncthreads();
    v16h a = load_A16x32_v(Alds, lane, LDA);
#pragma unroll
    for (int nt = 0; nt < 4; ++nt) {
      v16h b = load_Bk16(Wlds + (wave * 64 + nt * 16) * LDW, lane, LDW);
      acc[nt] = wmma16(a, b, acc[nt]);
    }
    __syncthreads();
  }

  int n0 = wave * 64 + (lane & 15);
  int mB = (lane & 16) ? 8 : 0;
#pragma unroll
  for (int nt = 0; nt < 4; ++nt)
#pragma unroll
    for (int r = 0; r < 8; ++r) {
      int col = n0 + nt * 16, row = mB + r;
      float v = acc[nt][r] + bias[col];
      Clds[row * 512 + col] = v;
      if (WRITE_RAW) rawOut[(size_t)(rowBase + row) * 512 + col] = v;
    }
  __syncthreads();
  ln_store(Clds, red, tid, rowBase, g, bb, outh);
}

// ---------------------------------------------------------------------------
// Kernel 2: plain f16 GEMM (no bias), 32-row blocks -- QKV projections.
// ---------------------------------------------------------------------------
__global__ __launch_bounds__(256) void qkv_kernel(
    const _Float16* __restrict__ A, const _Float16* __restrict__ Wt,
    _Float16* __restrict__ Out) {
  __shared__ __align__(16) _Float16 Alds[32 * LDA];
  __shared__ __align__(16) _Float16 Wlds[512 * LDW];
  int tid = threadIdx.x;
  int rowBase = blockIdx.x * 32;
  v8f acc[8];
#pragma unroll
  for (int i = 0; i < 8; ++i) acc[i] = v8zero();
  gemm_mainloop<512, 2>(A, rowBase, Wt, Alds, Wlds, acc, tid);
  int wave = tid >> 5, lane = tid & 31;
  int n0 = wave * 64 + (lane & 15);
  int mB = (lane & 16) ? 8 : 0;
#pragma unroll
  for (int m = 0; m < 2; ++m)
#pragma unroll
    for (int nt = 0; nt < 4; ++nt)
#pragma unroll
      for (int r = 0; r < 8; ++r)
        Out[(size_t)(rowBase + m * 16 + mB + r) * 512 + n0 + nt * 16] =
            (_Float16)acc[m * 4 + nt][r];
}

// ---------------------------------------------------------------------------
// Kernel 3: one attention problem per block (batch b, branch, head).
// ---------------------------------------------------------------------------
__global__ __launch_bounds__(256) void attn_kernel(
    const _Float16* __restrict__ qall, const _Float16* __restrict__ kall,
    const _Float16* __restrict__ vall, _Float16* __restrict__ ocat) {
  __shared__ __align__(16) _Float16 Qs[64 * LDQK];
  __shared__ __align__(16) _Float16 Ks[64 * LDQK];
  __shared__ __align__(16) _Float16 Vt[256 * LDVT];
  __shared__ __align__(16) _Float16 Ps[64 * LDP];
  __shared__ float Ss[64 * LDS_S];
  int tid = threadIdx.x;
  int b = blockIdx.x, br = blockIdx.y, head = blockIdx.z;
  size_t strideBr = (size_t)NROW * 512;
  const _Float16* Qg = qall + (size_t)br * strideBr;
  const _Float16* Kg = kall + (size_t)br * strideBr;
  const _Float16* Vg = vall + (size_t)br * strideBr;
  size_t base = ((size_t)b * 49) * 512 + (size_t)head * 256;

  for (int ci = tid; ci < 2048; ci += 256) {  // load Q/K/V (rows >= 49 zeroed)
    int row = ci >> 5;
    int c8  = (ci & 31) << 3;
    uint4 qv = make_uint4(0u, 0u, 0u, 0u), kv = qv, vv = qv;
    if (row < 49) {
      size_t off = base + (size_t)row * 512 + c8;
      qv = *(const uint4*)(Qg + off);
      kv = *(const uint4*)(Kg + off);
      vv = *(const uint4*)(Vg + off);
    }
    *(uint4*)(Qs + row * LDQK + c8) = qv;
    *(uint4*)(Ks + row * LDQK + c8) = kv;
    H8 u; u.q = vv;
#pragma unroll
    for (int j = 0; j < 8; ++j) Vt[(c8 + j) * LDVT + row] = u.h[j];  // transpose
  }
  __syncthreads();

  int wave = tid >> 5, lane = tid & 31;
  // S = Q K^T * scale: 4x4 tiles; each wave owns 2 tiles sharing one M block.
  v8f accS0 = v8zero(), accS1 = v8zero();
  int t0 = wave * 2;
  int mt0 = t0 >> 2, nt0 = t0 & 3, nt1 = (t0 + 1) & 3;
#pragma unroll
  for (int kc = 0; kc < 256; kc += 32) {
    v16h a  = load_A16x32_v(Qs + mt0 * 16 * LDQK + kc, lane, LDQK);
    v16h b0 = load_Bk16(Ks + nt0 * 16 * LDQK + kc, lane, LDQK);  // B^T of K
    accS0 = wmma16(a, b0, accS0);
    v16h b1 = load_Bk16(Ks + nt1 * 16 * LDQK + kc, lane, LDQK);
    accS1 = wmma16(a, b1, accS1);
  }
  int n  = lane & 15;
  int mB = (lane & 16) ? 8 : 0;
#pragma unroll
  for (int r = 0; r < 8; ++r) {
    Ss[(mt0 * 16 + mB + r) * LDS_S + nt0 * 16 + n] = accS0[r] * 0.0625f;  // 256^-0.5
    Ss[(mt0 * 16 + mB + r) * LDS_S + nt1 * 16 + n] = accS1[r] * 0.0625f;
  }
  __syncthreads();

  if (tid < 64) {  // per-row softmax over the 49 valid keys
    float mx = -3.0e38f;
    for (int c = 0; c < 49; ++c) mx = fmaxf(mx, Ss[tid * LDS_S + c]);
    float sum = 0.f;
    for (int c = 0; c < 49; ++c) {
      float e = __expf(Ss[tid * LDS_S + c] - mx);
      Ss[tid * LDS_S + c] = e;
      sum += e;
    }
    float inv = 1.f / sum;
    for (int c = 0; c < 49; ++c) Ps[tid * LDP + c] = (_Float16)(Ss[tid * LDS_S + c] * inv);
    for (int c = 49; c < 64; ++c) Ps[tid * LDP + c] = (_Float16)0.f;
  }
  __syncthreads();

  // O = P V : 4 x 16 tiles; each wave owns one M block and 8 N tiles.
  v8f accO[8];
#pragma unroll
  for (int i = 0; i < 8; ++i) accO[i] = v8zero();
  int mt = wave >> 1, ntBase = (wave & 1) * 8;
#pragma unroll
  for (int kc = 0; kc < 64; kc += 32) {
    v16h a = load_A16x32_v(Ps + mt * 16 * LDP + kc, lane, LDP);
#pragma unroll
    for (int i = 0; i < 8; ++i) {
      v16h bb = load_Bk16(Vt + (ntBase + i) * 16 * LDVT + kc, lane, LDVT);
      accO[i] = wmma16(a, bb, accO[i]);
    }
  }
  size_t obase = (size_t)b * 49 * 1536 + (size_t)br * 512 + (size_t)head * 256;
#pragma unroll
  for (int i = 0; i < 8; ++i) {
    int col = (ntBase + i) * 16 + n;
#pragma unroll
    for (int r = 0; r < 8; ++r) {
      int row = mt * 16 + mB + r;
      if (row < 49) ocat[obase + (size_t)row * 1536 + col] = (_Float16)accO[i][r];
    }
  }
}

// ---------------------------------------------------------------------------
// Kernel 4: x2 = x + concat(o)@Wo + bo ; xn5 = LN5(x2)  (16-row blocks)
// ---------------------------------------------------------------------------
__global__ __launch_bounds__(256) void wo_ln_kernel(
    const _Float16* __restrict__ Ocat, const _Float16* __restrict__ Wot,
    const float* __restrict__ bo, const float* __restrict__ xin,
    float* __restrict__ x2, const float* __restrict__ g5,
    const float* __restrict__ b5, _Float16* __restrict__ xn5) {
  __shared__ __align__(16) _Float16 Alds[16 * LDA];
  __shared__ __align__(16) _Float16 Wlds[512 * LDW];
  __shared__ float Clds[16 * 512];
  __shared__ float red[512];
  int tid = threadIdx.x;
  int rowBase = blockIdx.x * 16;
  v8f acc[4];
#pragma unroll
  for (int i = 0; i < 4; ++i) acc[i] = v8zero();
  gemm_mainloop<1536, 1>(Ocat, rowBase, Wot, Alds, Wlds, acc, tid);
  int wave = tid >> 5, lane = tid & 31;
  int n0 = wave * 64 + (lane & 15);
  int mB = (lane & 16) ? 8 : 0;
#pragma unroll
  for (int nt = 0; nt < 4; ++nt)
#pragma unroll
    for (int r = 0; r < 8; ++r) {
      int col = n0 + nt * 16, row = mB + r;
      size_t gi = (size_t)(rowBase + row) * 512 + col;
      float v = acc[nt][r] + bo[col] + xin[gi];
      Clds[row * 512 + col] = v;
      x2[gi] = v;
    }
  __syncthreads();
  ln_store(Clds, red, tid, rowBase, g5, b5, xn5);
}

// ---------------------------------------------------------------------------
// Kernel 5: h = relu(xn5 @ Wf1 + bf1), 32-row blocks, 4 column blocks.
// ---------------------------------------------------------------------------
__global__ __launch_bounds__(256) void ffn1_kernel(
    const _Float16* __restrict__ xn5, const _Float16* __restrict__ Wf1t,
    const float* __restrict__ bf1, _Float16* __restrict__ h) {
  __shared__ __align__(16) _Float16 Alds[32 * LDA];
  __shared__ __align__(16) _Float16 Wlds[512 * LDW];
  int tid = threadIdx.x;
  int rowBase = blockIdx.x * 32;
  int cb = blockIdx.y;  // 0..3
  v8f acc[8];
#pragma unroll
  for (int i = 0; i < 8; ++i) acc[i] = v8zero();
  gemm_mainloop<512, 2>(xn5, rowBase, Wf1t + (size_t)cb * 512 * 512, Alds, Wlds, acc, tid);
  int wave = tid >> 5, lane = tid & 31;
  int n0 = wave * 64 + (lane & 15);
  int mB = (lane & 16) ? 8 : 0;
#pragma unroll
  for (int m = 0; m < 2; ++m)
#pragma unroll
    for (int nt = 0; nt < 4; ++nt)
#pragma unroll
      for (int r = 0; r < 8; ++r) {
        int col = cb * 512 + n0 + nt * 16;
        float v = fmaxf(acc[m * 4 + nt][r] + bf1[col], 0.f);
        h[(size_t)(rowBase + m * 16 + mB + r) * 2048 + col] = (_Float16)v;
      }
}

// ---------------------------------------------------------------------------
// Kernel 6: out = x2 + h @ Wf2 + bf2, 32-row blocks, scatter to [B,512,7,7].
// ---------------------------------------------------------------------------
__global__ __launch_bounds__(256) void ffn2_kernel(
    const _Float16* __restrict__ h, const _Float16* __restrict__ Wf2t,
    const float* __restrict__ bf2, const float* __restrict__ x2,
    float* __restrict__ out) {
  __shared__ __align__(16) _Float16 Alds[32 * LDA];
  __shared__ __align__(16) _Float16 Wlds[512 * LDW];
  int tid = threadIdx.x;
  int rowBase = blockIdx.x * 32;
  v8f acc[8];
#pragma unroll
  for (int i = 0; i < 8; ++i) acc[i] = v8zero();
  gemm_mainloop<2048, 2>(h, rowBase, Wf2t, Alds, Wlds, acc, tid);
  int wave = tid >> 5, lane = tid & 31;
  int n0 = wave * 64 + (lane & 15);
  int mB = (lane & 16) ? 8 : 0;
#pragma unroll
  for (int m = 0; m < 2; ++m)
#pragma unroll
    for (int nt = 0; nt < 4; ++nt)
#pragma unroll
      for (int r = 0; r < 8; ++r) {
        int col = n0 + nt * 16;
        int rg = rowBase + m * 16 + mB + r;
        float v = acc[m * 4 + nt][r] + bf2[col] + x2[(size_t)rg * 512 + col];
        int b = rg / 49, t = rg % 49;
        out[((size_t)b * 512 + col) * 49 + t] = v;  // [B,512,7,7]
      }
}

// ---------------------------------------------------------------------------
// Host-side launch.
// ---------------------------------------------------------------------------
extern "C" void kernel_launch(void* const* d_in, const int* in_sizes, int n_in,
                              void* d_out, int out_size, void* d_ws, size_t ws_size,
                              hipStream_t stream) {
  (void)in_sizes; (void)n_in; (void)out_size; (void)ws_size;
  const float* c1 = (const float*)d_in[0];
  const float* c2 = (const float*)d_in[1];
  const float* c3 = (const float*)d_in[2];
  const float* c4 = (const float*)d_in[3];
  const float* Wp[4]  = {(const float*)d_in[4], (const float*)d_in[6],
                         (const float*)d_in[8], (const float*)d_in[10]};
  const float* bp[4]  = {(const float*)d_in[5], (const float*)d_in[7],
                         (const float*)d_in[9], (const float*)d_in[11]};
  const float* lng[5]; const float* lnb[5];
  for (int i = 0; i < 5; ++i) { lng[i] = (const float*)d_in[12 + 2 * i]; lnb[i] = (const float*)d_in[13 + 2 * i]; }
  const float* Wq[3]; const float* Wk[3]; const float* Wv[3];
  for (int br = 0; br < 3; ++br) {
    Wq[br] = (const float*)d_in[22 + 3 * br];
    Wk[br] = (const float*)d_in[23 + 3 * br];
    Wv[br] = (const float*)d_in[24 + 3 * br];
  }
  const float* Wo  = (const float*)d_in[31];
  const float* bo  = (const float*)d_in[32];
  const float* Wf1 = (const float*)d_in[33];
  const float* bf1 = (const float*)d_in[34];
  const float* Wf2 = (const float*)d_in[35];
  const float* bf2 = (const float*)d_in[36];

  char* ws = (char*)d_ws;
  const size_t sz_h = (size_t)NROW * DIMC * sizeof(_Float16);   // 12.85 MB
  _Float16* nbf0 = (_Float16*)(ws + 0 * sz_h);   // LN(e1)
  _Float16* nbf1 = (_Float16*)(ws + 1 * sz_h);   // LN(e2)
  _Float16* nbf2 = (_Float16*)(ws + 2 * sz_h);   // LN(e3)
  _Float16* nbf3 = (_Float16*)(ws + 3 * sz_h);   // LN(x)  (query side)
  _Float16* qb   = (_Float16*)(ws + 4 * sz_h);   // q[3]
  _Float16* kb   = (_Float16*)(ws + 7 * sz_h);   // k[3]
  _Float16* vb   = (_Float16*)(ws + 10 * sz_h);  // v[3]
  float*    xraw = (float*)(ws + 13 * sz_h);     // x (pre-LN, fp32 residual)
  _Float16* ocat = (_Float16*)(ws + 13 * sz_h + (size_t)NROW * DIMC * 4);
  char*     wend = ws + 13 * sz_h + (size_t)NROW * DIMC * 4 + (size_t)NROW * 1536 * 2;
  _Float16* wbase = (_Float16*)wend;             // transposed f16 weights
  // reuse regions dead after QKV/attention:
  float*    x2   = (float*)ws;                                   // over nbf0/nbf1
  _Float16* xn5  = (_Float16*)(ws + (size_t)NROW * DIMC * 4);    // over nbf2
  _Float16* hbuf = (_Float16*)(ws + 4 * sz_h);                   // over q/k

  dim3 blk(256);
  const int RB16 = NROW / 16;  // 784 blocks (16-row kernels)
  const int RB32 = NROW / 32;  // 392 blocks (32-row kernels)

  // 0) transpose all weights to f16 [N][K] once
  size_t wo_off = 0;
  auto prep = [&](const float* W, int K, int N) -> _Float16* {
    _Float16* p = wbase + wo_off;
    wo_off += (size_t)K * N;
    int total = K * N;
    wtrans_kernel<<<(total + 255) / 256, blk, 0, stream>>>(W, p, K, N);
    return p;
  };
  _Float16* Wpt[4];
  Wpt[0] = prep(Wp[0], 1024, 512);
  Wpt[1] = prep(Wp[1], 512, 512);
  Wpt[2] = prep(Wp[2], 256, 512);
  Wpt[3] = prep(Wp[3], 512, 512);
  _Float16 *Wqt[3], *Wkt[3], *Wvt[3];
  for (int br = 0; br < 3; ++br) {
    Wqt[br] = prep(Wq[br], 512, 512);
    Wkt[br] = prep(Wk[br], 512, 512);
    Wvt[br] = prep(Wv[br], 512, 512);
  }
  _Float16* Wot  = prep(Wo, 1536, 512);
  _Float16* Wf1t = prep(Wf1, 512, 2048);
  _Float16* Wf2t = prep(Wf2, 2048, 512);

  // 1) patchify + embed + LN (4 streams)
  embed_ln_kernel<64, 4, false><<<RB16, blk, 0, stream>>>(c1, Wpt[0], bp[0], lng[0], lnb[0], nbf0, nullptr);
  embed_ln_kernel<128, 2, false><<<RB16, blk, 0, stream>>>(c2, Wpt[1], bp[1], lng[1], lnb[1], nbf1, nullptr);
  embed_ln_kernel<256, 1, false><<<RB16, blk, 0, stream>>>(c3, Wpt[2], bp[2], lng[2], lnb[2], nbf2, nullptr);
  embed_ln_kernel<512, 1, true ><<<RB16, blk, 0, stream>>>(c4, Wpt[3], bp[3], lng[3], lnb[3], nbf3, xraw);

  // 2) QKV projections (no bias), 32-row blocks
  const _Float16* nbf[3] = {nbf0, nbf1, nbf2};
  for (int br = 0; br < 3; ++br) {
    qkv_kernel<<<RB32, blk, 0, stream>>>(nbf3,    Wqt[br], qb + (size_t)br * NROW * DIMC);
    qkv_kernel<<<RB32, blk, 0, stream>>>(nbf[br], Wkt[br], kb + (size_t)br * NROW * DIMC);
    qkv_kernel<<<RB32, blk, 0, stream>>>(nbf[br], Wvt[br], vb + (size_t)br * NROW * DIMC);
  }

  // 3) attention: (batch=256, branch=3, head=2)
  attn_kernel<<<dim3(256, 3, 2), blk, 0, stream>>>(qb, kb, vb, ocat);

  // 4) output projection + residual + LN5
  wo_ln_kernel<<<RB16, blk, 0, stream>>>(ocat, Wot, bo, xraw, x2, lng[4], lnb[4], xn5);

  // 5) FFN up + relu (32-row blocks)
  ffn1_kernel<<<dim3(RB32, 4), blk, 0, stream>>>(xn5, Wf1t, bf1, hbuf);

  // 6) FFN down + residual + transpose scatter to [B,512,7,7] (32-row blocks)
  ffn2_kernel<<<RB32, blk, 0, stream>>>(hbuf, Wf2t, bf2, x2, (float*)d_out);
}
